// LSTM_71090298683742
// MI455X (gfx1250) — matrix operand order repaired and checked by reading, not compiled, not run on hardware
//
#include <hip/hip_runtime.h>
#include <hip/hip_bf16.h>

// ---------------------------------------------------------------------------
// LSTM forward on gfx1250 (CDNA5, wave32, WMMA)
//   Phase 0: f32 -> bf16 conversion of x, Wi, Wh
//   Phase 1: x_pre = x @ Wi^T + bi   (WMMA bf16 GEMM, f32 accumulate,
//            double-buffered LDS: 1 barrier / K-step, loads overlap WMMA)
//   Phase 2: persistent recurrent kernel, Wh slice resident in LDS,
//            h/c in registers, ping-pong bf16 h buffer + grid barrier,
//            next-step x_pre prefetched during the barrier window
// ---------------------------------------------------------------------------

typedef __attribute__((ext_vector_type(16))) __bf16 v16bf;
typedef __attribute__((ext_vector_type(8)))  float  v8f;

#define Bz   64
#define Tz   1024
#define Iz   512
#define Hz   512
#define G4H  2048           // 4*H
#define KD   512            // K dimension (== I == H)

union Frag32 { uint4 u[2]; v16bf v; };

static __device__ inline v16bf load_frag(const unsigned short* p, int second_off_elems) {
    Frag32 f;
    f.u[0] = *(const uint4*)(p);
    f.u[1] = *(const uint4*)(p + second_off_elems);
    return f.v;
}

static __device__ inline unsigned short f32_to_bf16_rne(float x) {
    union { float f; unsigned int u; } v; v.f = x;
    unsigned int r = v.u + 0x7FFFu + ((v.u >> 16) & 1u);
    return (unsigned short)(r >> 16);
}

static __device__ inline float fsigmoid(float x) {
    return 1.0f / (1.0f + __expf(-x));
}
static __device__ inline float ftanh(float x) {
    return 2.0f / (1.0f + __expf(-2.0f * x)) - 1.0f;
}

static __device__ inline v8f v8f_zero() {
    return (v8f){0.f, 0.f, 0.f, 0.f, 0.f, 0.f, 0.f, 0.f};
}

// ---------------------------------------------------------------------------
// Phase 0: elementwise f32 -> bf16
// ---------------------------------------------------------------------------
__global__ void cvt_bf16_kernel(const float* __restrict__ src,
                                unsigned short* __restrict__ dst, int n) {
    int i = blockIdx.x * blockDim.x + threadIdx.x;
    if (i < n) dst[i] = f32_to_bf16_rne(src[i]);
}

// ---------------------------------------------------------------------------
// Phase 1: x_pre[M=B*T, N=4H] = Xb[M,K] * Wib[N,K]^T + bi[N]
// Workgroup tile 128x128, BK=32, 256 threads = 8 waves (4 M x 2 N),
// each wave: 2x4 accumulator tiles of 16x16.
// Double-buffered LDS: one s_barrier per K-step; next tile's global loads
// are issued right after the barrier so they overlap the 8-WMMA burst.
// ---------------------------------------------------------------------------
__global__ __launch_bounds__(256)
void lstm_xpre_gemm(const unsigned short* __restrict__ Xb,
                    const unsigned short* __restrict__ Wib,
                    const float* __restrict__ bi,
                    float* __restrict__ xpre) {
    __shared__ unsigned short sA[2][128 * 32];   // 2 x 8 KB
    __shared__ unsigned short sB[2][128 * 32];   // 2 x 8 KB

    const int tid   = threadIdx.x;
    const int lane  = tid & 31;
    const int wave  = tid >> 5;
    const int tileM = blockIdx.x * 128;          // 512 tiles
    const int tileN = blockIdx.y * 128;          // 16 tiles
    const int waveM = (wave & 3) * 32;           // 0/32/64/96
    const int waveN = (wave >> 2) * 64;          // 0/64

    const int lrow = lane & 15;
    const int aK0  = (lane < 16) ? 0 : 8;        // A 16x32 bf16 layout
    const int bK0  = (lane < 16) ? 0 : 8;        // B 32x16 bf16 layout

    v8f acc[2][4];
    for (int i = 0; i < 2; ++i)
        for (int j = 0; j < 4; ++j) acc[i][j] = v8f_zero();

    // staging: 2 threads per row, 16 bf16 (32 B) each
    const int srow = tid >> 1, shalf = tid & 1;
    const unsigned short* gA = Xb  + (size_t)(tileM + srow) * KD + shalf * 16;
    const unsigned short* gB = Wib + (size_t)(tileN + srow) * KD + shalf * 16;
    const int sOff = srow * 32 + shalf * 16;

    uint4 ra0 = ((const uint4*)gA)[0], ra1 = ((const uint4*)gA)[1];
    uint4 rb0 = ((const uint4*)gB)[0], rb1 = ((const uint4*)gB)[1];

    for (int kt = 0; kt < KD / 32; ++kt) {
        const int buf = kt & 1;
        {   // commit staged tile kt
            uint4* da = (uint4*)(sA[buf] + sOff);
            da[0] = ra0; da[1] = ra1;
            uint4* db = (uint4*)(sB[buf] + sOff);
            db[0] = rb0; db[1] = rb1;
        }
        __syncthreads();

        if (kt + 1 < KD / 32) {  // start fetching tile kt+1 (overlaps WMMA)
            const uint4* na = (const uint4*)(gA + (kt + 1) * 32);
            ra0 = na[0]; ra1 = na[1];
            const uint4* nb = (const uint4*)(gB + (kt + 1) * 32);
            rb0 = nb[0]; rb1 = nb[1];
        }

        v16bf afrag[2], bfrag[4];
        for (int i = 0; i < 2; ++i) {
            const unsigned short* p = sA[buf] + (waveM + i * 16 + lrow) * 32 + aK0;
            afrag[i] = load_frag(p, 16);   // K chunks {0..7,16..23} / {8..15,24..31}
        }
        for (int j = 0; j < 4; ++j) {
            const unsigned short* p = sB[buf] + (waveN + j * 16 + lrow) * 32 + bK0;
            bfrag[j] = load_frag(p, 16);   // K chunks {0..7,16..23} / {8..15,24..31}
        }
        for (int i = 0; i < 2; ++i)
            for (int j = 0; j < 4; ++j)
                acc[i][j] = __builtin_amdgcn_wmma_f32_16x16x32_bf16(
                    false, afrag[i], false, bfrag[j], (short)0, acc[i][j],
                    false, false);
    }

    // epilogue: C/D layout -> lane holds col = lrow, rows {0..7}+8*(lane/16)
    const int rbase = (lane < 16) ? 0 : 8;
    for (int i = 0; i < 2; ++i)
        for (int j = 0; j < 4; ++j) {
            int col  = tileN + waveN + j * 16 + lrow;
            int row0 = tileM + waveM + i * 16 + rbase;
            float bv = bi[col];
            for (int r = 0; r < 8; ++r)
                xpre[(size_t)(row0 + r) * G4H + col] = acc[i][j][r] + bv;
        }
}

// ---------------------------------------------------------------------------
// Phase 2: persistent recurrent kernel. 32 WGs x 128 threads (4 waves).
// WG w owns h-dims [16w,16w+16). Its Wh slice (4 gates x 16 rows x 512 K,
// bf16 = 64 KB) is staged once into LDS and reused for all 1024 steps.
// Wave m handles batches [16m,16m+16). Lane's 4 gate accumulators cover the
// SAME (batch,hdim) cells -> gate math is lane-local; h,c live in registers.
// h exchanged via ping-pong bf16 global buffer + monotonic atomic barrier.
// ---------------------------------------------------------------------------
__global__ __launch_bounds__(128)
void lstm_recurrent(const float* __restrict__ xpre,
                    const unsigned short* __restrict__ Whb,
                    const float* __restrict__ bh,
                    const int* __restrict__ lens,
                    unsigned short* __restrict__ hbuf0,
                    unsigned short* __restrict__ hbuf1,
                    unsigned int* __restrict__ bar,
                    float* __restrict__ out) {
    __shared__ unsigned short sW[64 * KD];    // 64 KB, persistent Wh slice

    const int tid  = threadIdx.x;
    const int lane = tid & 31;
    const int wave = tid >> 5;                // m-tile (batch group)
    const int hc0  = blockIdx.x * 16;         // owned h-dim base

    // stage Wh slice: LDS row r = gate*16 + j  <-  Wh row gate*512 + hc0 + j
    for (int i = tid; i < 64 * (KD / 8); i += 128) {
        int r = i >> 6, cq = i & 63;
        int sr = (r >> 4) * Hz + hc0 + (r & 15);
        ((uint4*)sW)[(size_t)r * (KD / 8) + cq] =
            ((const uint4*)(Whb + (size_t)sr * KD))[cq];
    }
    __syncthreads();

    const int lrow  = lane & 15;
    const int rbase = (lane < 16) ? 0 : 8;
    const int aK0   = (lane < 16) ? 0 : 8;
    const int bK0   = (lane < 16) ? 0 : 8;
    const int hdim  = hc0 + lrow;
    const int mrow0 = wave * 16 + rbase;      // first of this lane's 8 batches

    float hreg[8], creg[8];
    int   lenv[8];
    for (int r = 0; r < 8; ++r) {
        hreg[r] = 0.f; creg[r] = 0.f;
        lenv[r] = lens[mrow0 + r];
    }
    float bh_g[4];
    for (int g = 0; g < 4; ++g) bh_g[g] = bh[g * Hz + hdim];

    unsigned short* hb[2] = {hbuf0, hbuf1};

    for (int t = 0; t < Tz; ++t) {
        const unsigned short* hr = hb[t & 1];
        unsigned short*       hw = hb[(t + 1) & 1];

        v8f acc[4];
        for (int g = 0; g < 4; ++g) acc[g] = v8f_zero();

        for (int k0 = 0; k0 < KD; k0 += 32) {
            const unsigned short* pa =
                hr + (size_t)(wave * 16 + lrow) * KD + k0 + aK0;
            v16bf af = load_frag(pa, 16);
            for (int g = 0; g < 4; ++g) {
                const unsigned short* pb =
                    sW + (size_t)(g * 16 + lrow) * KD + k0 + bK0;
                v16bf bf = load_frag(pb, 16);
                acc[g] = __builtin_amdgcn_wmma_f32_16x16x32_bf16(
                    false, af, false, bf, (short)0, acc[g], false, false);
            }
        }

        for (int r = 0; r < 8; ++r) {
            int b = mrow0 + r;
            size_t base = ((size_t)b * Tz + t) * (size_t)G4H;
            float pi = acc[0][r] + xpre[base + 0 * Hz + hdim] + bh_g[0];
            float pf = acc[1][r] + xpre[base + 1 * Hz + hdim] + bh_g[1];
            float po = acc[2][r] + xpre[base + 2 * Hz + hdim] + bh_g[2];
            float pg = acc[3][r] + xpre[base + 3 * Hz + hdim] + bh_g[3];
            float ig = fsigmoid(pi), fg = fsigmoid(pf), og = fsigmoid(po);
            float gg = ftanh(pg);
            float cn = creg[r] * fg + ig * gg;
            float hn = og * ftanh(cn);
            bool  m  = (t < lenv[r]);
            creg[r] = m ? cn : creg[r];
            hreg[r] = m ? hn : hreg[r];
            out[((size_t)b * Tz + t) * (size_t)Hz + hdim] = m ? hn : 0.0f;
            hw[(size_t)b * Hz + hdim] = f32_to_bf16_rne(hreg[r]);
        }

        // pull next step's x_pre gate rows toward the WGP while we sync
        if (t + 1 < Tz) {
            for (int r = 0; r < 8; ++r) {
                size_t bn = ((size_t)(mrow0 + r) * Tz + (t + 1)) * (size_t)G4H;
                for (int g = 0; g < 4; ++g)
                    __builtin_prefetch(&xpre[bn + g * Hz + hdim], 0, 1);
            }
        }

        // grid-wide barrier: monotonic counter, target = 32*(t+1)
        __threadfence();
        __syncthreads();
        if (tid == 0) {
            __hip_atomic_fetch_add(bar, 1u, __ATOMIC_ACQ_REL,
                                   __HIP_MEMORY_SCOPE_AGENT);
            unsigned int tgt = 32u * (unsigned int)(t + 1);
            while (__hip_atomic_load(bar, __ATOMIC_ACQUIRE,
                                     __HIP_MEMORY_SCOPE_AGENT) < tgt) {
                __builtin_amdgcn_s_sleep(1);
            }
        }
        __syncthreads();
    }

    // final hT / cT (f32) appended after outputs
    float* hT = out + (size_t)Bz * Tz * Hz;
    float* cT = hT + (size_t)Bz * Hz;
    for (int r = 0; r < 8; ++r) {
        int b = mrow0 + r;
        hT[(size_t)b * Hz + hdim] = hreg[r];
        cT[(size_t)b * Hz + hdim] = creg[r];
    }
}

// ---------------------------------------------------------------------------
extern "C" void kernel_launch(void* const* d_in, const int* in_sizes, int n_in,
                              void* d_out, int out_size, void* d_ws, size_t ws_size,
                              hipStream_t stream) {
    const float* x    = (const float*)d_in[0];
    const int*   lens = (const int*)d_in[1];
    const float* Wi   = (const float*)d_in[2];
    const float* bi   = (const float*)d_in[3];
    const float* Wh   = (const float*)d_in[4];
    const float* bh   = (const float*)d_in[5];
    float*       out  = (float*)d_out;

    char* ws = (char*)d_ws;
    size_t off = 0;
    unsigned short* xb   = (unsigned short*)(ws + off); off += (size_t)Bz * Tz * Iz * 2;
    unsigned short* wib  = (unsigned short*)(ws + off); off += (size_t)G4H * Iz * 2;
    unsigned short* whb  = (unsigned short*)(ws + off); off += (size_t)G4H * Hz * 2;
    float*          xpre = (float*)(ws + off);          off += (size_t)Bz * Tz * G4H * 4;
    unsigned short* h0   = (unsigned short*)(ws + off); off += (size_t)Bz * Hz * 2;
    unsigned short* h1   = (unsigned short*)(ws + off); off += (size_t)Bz * Hz * 2;
    unsigned int*   bar  = (unsigned int*)(ws + off);   off += 256;

    // zero h ping-pong buffers + barrier counter (contiguous region)
    hipMemsetAsync(h0, 0, (size_t)Bz * Hz * 2 * 2 + 256, stream);

    // Phase 0: conversions
    const int n_x = Bz * Tz * Iz;
    cvt_bf16_kernel<<<(n_x + 255) / 256, 256, 0, stream>>>(x, xb, n_x);
    const int n_w = G4H * Iz;
    cvt_bf16_kernel<<<(n_w + 255) / 256, 256, 0, stream>>>(Wi, wib, n_w);
    cvt_bf16_kernel<<<(n_w + 255) / 256, 256, 0, stream>>>(Wh, whb, n_w);

    // Phase 1: x_pre GEMM  (M=65536 / 128, N=2048 / 128)
    dim3 g1((Bz * Tz) / 128, G4H / 128);
    lstm_xpre_gemm<<<g1, 256, 0, stream>>>(xb, wib, bi, xpre);

    // Phase 2: persistent recurrent kernel (32 WGs x 128 threads)
    lstm_recurrent<<<Hz / 16, 128, 0, stream>>>(xpre, whb, bh, lens,
                                                h0, h1, bar, out);
}
